// GraphModel_73667279061369
// MI455X (gfx1250) — matrix operand order repaired and checked by reading, not compiled, not run on hardware
//
#include <hip/hip_runtime.h>
#include <math.h>

// ---------------------------------------------------------------------------
// Problem constants (match setup_inputs: G=64, M=256, D=256, E=256, k=16)
// ---------------------------------------------------------------------------
#define G_   64
#define M_   256
#define D_   256
#define E_   256
#define KNN  16
#define GM   16384          // G*M rows per input
#define NUM_SINK 8
#define MASS 0.9f

#define LDA1 520            // 512 + 8 pad (bank-conflict avoidance)
#define LDC1 264            // 256 + 8 pad
#define LDX  264

typedef __bf16 bf16;
typedef bf16  bf16x16 __attribute__((ext_vector_type(16)));
typedef float floatx8 __attribute__((ext_vector_type(8)));

__device__ __forceinline__ floatx8 zero8() {
  floatx8 z = {0.f, 0.f, 0.f, 0.f, 0.f, 0.f, 0.f, 0.f};
  return z;
}

// D = A(16x32) * B(32x16) + C, bf16 inputs, f32 accumulate
__device__ __forceinline__ floatx8 wmma_bf(bf16x16 a, bf16x16 b, floatx8 c) {
  return __builtin_amdgcn_wmma_f32_16x16x32_bf16(
      /*neg_a=*/false, a, /*neg_b=*/false, b,
      /*c_mod=*/(short)0, c, /*reuse_a=*/false, /*reuse_b=*/false);
}

// A-fragment: 16x32 (MxK) row-major source, stride ld elements.
// Layout: lanes 0-15 -> K 0..15, lanes 16-31 -> K 16..31 (matches B),
// VGPR v holds K pair (2 bf16).
__device__ __forceinline__ bf16x16 load_A_frag(const bf16* src, int ld) {
  const int lane = threadIdx.x & 31;
  const int m  = lane & 15;
  const int kb = (lane >> 4) << 4;          // 0 or 16
  bf16x16 a;
#pragma unroll
  for (int v = 0; v < 8; ++v) {
    const int k = kb + (v << 1);  // base + pair
    a[2 * v]     = src[m * ld + k];
    a[2 * v + 1] = src[m * ld + k + 1];
  }
  return a;
}

// B-fragment: 32x16 (KxN), loaded from a TRANSPOSED (NxK row-major) source so
// each lane reads 16 contiguous bf16 (32B). Layout used here (contiguous, same as A): lanes 0-15 K 0..15,
// lanes 16-31 K 16..31, element i of the vector = K kb+i.
__device__ __forceinline__ bf16x16 load_B_fragT(const bf16* srcT, int ld) {
  const int lane = threadIdx.x & 31;
  const int n  = lane & 15;
  const int kb = (lane >> 4) << 4;          // 0 or 16
  bf16x16 b;
#pragma unroll
  for (int i = 0; i < 16; ++i) b[i] = srcT[n * ld + kb + i];
  return b;
}

__device__ __forceinline__ float wave_sum(float v) {
#pragma unroll
  for (int o = 16; o > 0; o >>= 1) v += __shfl_xor(v, o, 32);
  return v;
}
__device__ __forceinline__ float wave_max(float v) {
#pragma unroll
  for (int o = 16; o > 0; o >>= 1) v = fmaxf(v, __shfl_xor(v, o, 32));
  return v;
}

// ---------------------------------------------------------------------------
// 0) f32 [K][N] -> bf16 transposed [N][K] (weights; tiny, one-shot)
// ---------------------------------------------------------------------------
__global__ void k_cvtT(const float* __restrict__ src, bf16* __restrict__ dstT,
                       int K, int N) {
  const int i = blockIdx.x * 256 + threadIdx.x;
  if (i < K * N) {
    const int k = i / N, n = i - k * N;
    dstT[(size_t)n * K + k] = (bf16)src[i];
  }
}

// ---------------------------------------------------------------------------
// 1) Encoder: h = relu(x @ W_enc + b_enc), bf16 out. 1 block = 16 rows.
//    grid.x = 2 inputs * 1024 row-tiles
// ---------------------------------------------------------------------------
__global__ void k_encoder(const float* __restrict__ xa, const float* __restrict__ xb,
                          const bf16* __restrict__ WencT, const float* __restrict__ benc,
                          bf16* __restrict__ hbf) {
  __shared__ bf16 Ax[16 * LDX];
  const int b     = blockIdx.x;
  const int input = b >> 10;
  const int rt    = b & 1023;
  const float* x  = (input == 0) ? xa : xb;
  const size_t row0 = (size_t)input * GM + (size_t)rt * 16;
  const int t = threadIdx.x;
#pragma unroll 4
  for (int r = 0; r < 16; ++r)
    Ax[r * LDX + t] = (bf16)x[((size_t)rt * 16 + r) * D_ + t];
  __syncthreads();

  const int wid = t >> 5, lane = t & 31;
  const int nlane = lane & 15, mb = (lane >> 4) << 3;
  const int nt0 = wid * 2, nt1 = nt0 + 1;
  floatx8 acc0 = zero8(), acc1 = zero8();
#pragma unroll
  for (int kt = 0; kt < 8; ++kt) {
    bf16x16 a  = load_A_frag(Ax + kt * 32, LDX);
    bf16x16 b0 = load_B_fragT(WencT + (size_t)(nt0 * 16) * D_ + kt * 32, D_);
    bf16x16 b1 = load_B_fragT(WencT + (size_t)(nt1 * 16) * D_ + kt * 32, D_);
    acc0 = wmma_bf(a, b0, acc0);
    acc1 = wmma_bf(a, b1, acc1);
  }
  const int n0 = nt0 * 16 + nlane, n1 = nt1 * 16 + nlane;
  const float bb0 = benc[n0], bb1 = benc[n1];
#pragma unroll
  for (int v = 0; v < 8; ++v) {
    const size_t r = row0 + mb + v;
    hbf[r * D_ + n0] = (bf16)fmaxf(acc0[v] + bb0, 0.f);
    hbf[r * D_ + n1] = (bf16)fmaxf(acc1[v] + bb1, 0.f);
  }
}

// ---------------------------------------------------------------------------
// 2) Row squared norms of h (wave per row)
// ---------------------------------------------------------------------------
__global__ void k_sq(const bf16* __restrict__ hbf, float* __restrict__ sqv) {
  const int wid = threadIdx.x >> 5, lane = threadIdx.x & 31;
  const size_t row = (size_t)blockIdx.x * 8 + wid;
  const bf16* p = hbf + row * D_;
  float s = 0.f;
#pragma unroll
  for (int j = 0; j < 8; ++j) { const float v = (float)p[lane + 32 * j]; s += v * v; }
  s = wave_sum(s);
  if (lane == 0) sqv[row] = s;
}

// ---------------------------------------------------------------------------
// 3) Gram + pairwise distance: d2 = sq[m]+sq[n]-2 h.hT (+1e9 diag)
//    grid.x = 2 inputs * 64 graphs * 16 row-tiles
// ---------------------------------------------------------------------------
__global__ void k_gram(const bf16* __restrict__ hbf, const float* __restrict__ sqv,
                       float* __restrict__ d2v) {
  const int b = blockIdx.x;
  const int input = b >> 10, g = (b >> 4) & 63, rt = b & 15;
  const size_t gbase = (size_t)input * GM + (size_t)g * M_;
  const bf16* hg = hbf + gbase * D_;
  const int t = threadIdx.x, wid = t >> 5, lane = t & 31;
  const int nlane = lane & 15, mb = (lane >> 4) << 3;
  const int nt0 = wid * 2, nt1 = nt0 + 1;
  floatx8 acc0 = zero8(), acc1 = zero8();
#pragma unroll
  for (int kt = 0; kt < 8; ++kt) {
    bf16x16 a  = load_A_frag(hg + (size_t)(rt * 16) * D_ + kt * 32, D_);
    bf16x16 b0 = load_B_fragT(hg + (size_t)(nt0 * 16) * D_ + kt * 32, D_);
    bf16x16 b1 = load_B_fragT(hg + (size_t)(nt1 * 16) * D_ + kt * 32, D_);
    acc0 = wmma_bf(a, b0, acc0);
    acc1 = wmma_bf(a, b1, acc1);
  }
#pragma unroll
  for (int s = 0; s < 2; ++s) {
    const floatx8 acc = s ? acc1 : acc0;
    const int n = (s ? nt1 : nt0) * 16 + nlane;
    const float sqn = sqv[gbase + n];
#pragma unroll
    for (int v = 0; v < 8; ++v) {
      const int m = rt * 16 + mb + v;
      float d2 = sqv[gbase + m] + sqn - 2.f * acc[v];
      if (m == n) d2 += 1e9f;
      d2v[(gbase + m) * M_ + n] = d2;
    }
  }
}

// ---------------------------------------------------------------------------
// 4) top-k (k=16) smallest d2 per row (wave per row, deterministic ties)
// ---------------------------------------------------------------------------
__global__ void k_topk(const float* __restrict__ d2v, int* __restrict__ idxv) {
  const int wid = threadIdx.x >> 5, lane = threadIdx.x & 31;
  const size_t row = (size_t)blockIdx.x * 8 + wid;
  const float* p = d2v + row * M_;
  float v[8];
#pragma unroll
  for (int j = 0; j < 8; ++j) v[j] = p[lane + 32 * j];
#pragma unroll 1
  for (int it = 0; it < KNN; ++it) {
    float bv = 3.0e38f; int bi = M_;
#pragma unroll
    for (int j = 0; j < 8; ++j)
      if (v[j] < bv) { bv = v[j]; bi = lane + 32 * j; }
#pragma unroll
    for (int o = 16; o > 0; o >>= 1) {
      const float ov = __shfl_xor(bv, o, 32);
      const int   oi = __shfl_xor(bi, o, 32);
      if (ov < bv || (ov == bv && oi < bi)) { bv = ov; bi = oi; }
    }
    if (lane == 0) idxv[row * KNN + it] = bi;
#pragma unroll
    for (int j = 0; j < 8; ++j)
      if (bi == lane + 32 * j) v[j] = 3.0e38f;
  }
}

// ---------------------------------------------------------------------------
// 5) Fused edge MLP: per node build e=[h, nbr-h] (16x512 bf16 in LDS),
//    GEMM1 (x W1, relu) -> GEMM2 (x W2) -> max over the 16 neighbors.
//    Outputs bf16 conv row + its L2 norm. One block (8 waves) per node.
// ---------------------------------------------------------------------------
__global__ void k_edge(const bf16* __restrict__ hbf, const int* __restrict__ idxv,
                       const bf16* __restrict__ W1T, const float* __restrict__ b1,
                       const bf16* __restrict__ W2T, const float* __restrict__ b2,
                       bf16* __restrict__ convbf, float* __restrict__ nrm) {
  __shared__ bf16 A1[16 * LDA1];   // [k=16][2D=512] edge features
  __shared__ bf16 C1[16 * LDC1];   // [k=16][E=256]  relu(e@W1+b1)
  __shared__ float hval[D_];
  __shared__ float wred[8];

  const int blk = blockIdx.x;
  const int input = blk >> 14;
  const int rem = blk & 16383;
  const int g = rem >> 8, m = rem & 255;
  const size_t node = (size_t)input * GM + (size_t)g * M_ + m;
  const bf16* hg = hbf + ((size_t)input * GM + (size_t)g * M_) * D_;
  const int* nb = idxv + node * KNN;

  const int t = threadIdx.x;
  const float cen = (float)hg[(size_t)m * D_ + t];
#pragma unroll 4
  for (int j = 0; j < KNN; ++j) {
    const int nj = nb[j];
    const float nv = (float)hg[(size_t)nj * D_ + t];
    A1[j * LDA1 + t]      = (bf16)cen;
    A1[j * LDA1 + D_ + t] = (bf16)(nv - cen);
  }
  __syncthreads();

  const int wid = t >> 5, lane = t & 31;
  const int nlane = lane & 15, mb = (lane >> 4) << 3;
  const int nt0 = wid * 2, nt1 = nt0 + 1;

  // ---- GEMM1: [16,512] @ W1 -> [16,256], relu, bf16 to LDS ----
  {
    floatx8 acc0 = zero8(), acc1 = zero8();
#pragma unroll 4
    for (int kt = 0; kt < 16; ++kt) {
      bf16x16 a  = load_A_frag(A1 + kt * 32, LDA1);
      bf16x16 bb0 = load_B_fragT(W1T + (size_t)(nt0 * 16) * (2 * D_) + kt * 32, 2 * D_);
      bf16x16 bb1 = load_B_fragT(W1T + (size_t)(nt1 * 16) * (2 * D_) + kt * 32, 2 * D_);
      acc0 = wmma_bf(a, bb0, acc0);
      acc1 = wmma_bf(a, bb1, acc1);
    }
#pragma unroll
    for (int s = 0; s < 2; ++s) {
      const floatx8 acc = s ? acc1 : acc0;
      const int n = (s ? nt1 : nt0) * 16 + nlane;
      const float bv = b1[n];
#pragma unroll
      for (int v = 0; v < 8; ++v)
        C1[(mb + v) * LDC1 + n] = (bf16)fmaxf(acc[v] + bv, 0.f);
    }
  }
  __syncthreads();

  // ---- GEMM2: [16,256] @ W2 -> [16,256], + b2, column-max over k ----
  {
    floatx8 acc0 = zero8(), acc1 = zero8();
#pragma unroll
    for (int kt = 0; kt < 8; ++kt) {
      bf16x16 a  = load_A_frag(C1 + kt * 32, LDC1);
      bf16x16 bb0 = load_B_fragT(W2T + (size_t)(nt0 * 16) * E_ + kt * 32, E_);
      bf16x16 bb1 = load_B_fragT(W2T + (size_t)(nt1 * 16) * E_ + kt * 32, E_);
      acc0 = wmma_bf(a, bb0, acc0);
      acc1 = wmma_bf(a, bb1, acc1);
    }
#pragma unroll
    for (int s = 0; s < 2; ++s) {
      const floatx8 acc = s ? acc1 : acc0;
      const int n = (s ? nt1 : nt0) * 16 + nlane;
      const float bv = b2[n];
      float cmax = -3.0e38f;
#pragma unroll
      for (int v = 0; v < 8; ++v) cmax = fmaxf(cmax, acc[v] + bv);
      cmax = fmaxf(cmax, __shfl_xor(cmax, 16, 32));   // other 8 neighbor rows
      if (lane < 16) hval[n] = cmax;
    }
  }
  __syncthreads();

  const float hv = hval[t];
  convbf[node * D_ + t] = (bf16)hv;
  const float ss = wave_sum(hv * hv);
  if (lane == 0) wred[wid] = ss;
  __syncthreads();
  if (t == 0) {
    float tot = 0.f;
#pragma unroll
    for (int w = 0; w < 8; ++w) tot += wred[w];
    nrm[node] = sqrtf(tot);
  }
}

// ---------------------------------------------------------------------------
// 6) corr = tra @ detT ; log_p0 = corr / (n1*n2*lam)
//    grid.x = 64 graphs * 16 row-tiles
// ---------------------------------------------------------------------------
__global__ void k_corr(const bf16* __restrict__ convbf, const float* __restrict__ nrm,
                       const float* __restrict__ eps, float* __restrict__ lp) {
  const int b = blockIdx.x;
  const int g = b >> 4, rt = b & 15;
  const bf16* tra = convbf + (size_t)g * M_ * D_;
  const bf16* det = convbf + ((size_t)GM + (size_t)g * M_) * D_;
  const float lam = expf(eps[0]) + 0.03f;
  const int t = threadIdx.x, wid = t >> 5, lane = t & 31;
  const int nlane = lane & 15, mb = (lane >> 4) << 3;
  const int nt0 = wid * 2, nt1 = nt0 + 1;
  floatx8 acc0 = zero8(), acc1 = zero8();
#pragma unroll
  for (int kt = 0; kt < 8; ++kt) {
    bf16x16 a  = load_A_frag(tra + (size_t)(rt * 16) * D_ + kt * 32, D_);
    bf16x16 b0 = load_B_fragT(det + (size_t)(nt0 * 16) * D_ + kt * 32, D_);
    bf16x16 b1 = load_B_fragT(det + (size_t)(nt1 * 16) * D_ + kt * 32, D_);
    acc0 = wmma_bf(a, b0, acc0);
    acc1 = wmma_bf(a, b1, acc1);
  }
#pragma unroll
  for (int s = 0; s < 2; ++s) {
    const floatx8 acc = s ? acc1 : acc0;
    const int n = (s ? nt1 : nt0) * 16 + nlane;
    const float n2 = nrm[GM + g * M_ + n];
#pragma unroll
    for (int v = 0; v < 8; ++v) {
      const int mm = rt * 16 + mb + v;
      const float n1 = nrm[g * M_ + mm];
      lp[(size_t)g * M_ * M_ + (size_t)mm * M_ + n] = acc[v] / (n1 * n2 * lam);
    }
  }
}

// ---------------------------------------------------------------------------
// 7) Sinkhorn: log-domain row / column normalization (wave per row/col)
// ---------------------------------------------------------------------------
__global__ void k_sink_row(float* __restrict__ lp, float* __restrict__ out,
                           int final_pass) {
  const int wid = threadIdx.x >> 5, lane = threadIdx.x & 31;
  const size_t row = (size_t)blockIdx.x * 8 + wid;
  float* p = lp + row * M_;
  float v[8];
#pragma unroll
  for (int j = 0; j < 8; ++j) v[j] = p[lane + 32 * j];
  float mx = -3.0e38f;
#pragma unroll
  for (int j = 0; j < 8; ++j) mx = fmaxf(mx, v[j]);
  mx = wave_max(mx);
  float s = 0.f;
#pragma unroll
  for (int j = 0; j < 8; ++j) s += expf(v[j] - mx);
  s = wave_sum(s);
  const float delta = (mx + logf(s)) - logf(MASS);
  if (final_pass) {
    float* o = out + row * M_;
#pragma unroll
    for (int j = 0; j < 8; ++j) o[lane + 32 * j] = expf(v[j] - delta);
  } else {
#pragma unroll
    for (int j = 0; j < 8; ++j) p[lane + 32 * j] = v[j] - delta;
  }
}

__global__ void k_sink_col(float* __restrict__ lp) {
  const int wid = threadIdx.x >> 5, lane = threadIdx.x & 31;
  const int col = blockIdx.x * 8 + wid;
  const int g = col >> 8, n = col & 255;
  float* p = lp + (size_t)g * M_ * M_ + n;
  float v[8];
#pragma unroll
  for (int j = 0; j < 8; ++j) v[j] = p[(size_t)(lane + 32 * j) * M_];
  float mx = -3.0e38f;
#pragma unroll
  for (int j = 0; j < 8; ++j) mx = fmaxf(mx, v[j]);
  mx = wave_max(mx);
  float s = 0.f;
#pragma unroll
  for (int j = 0; j < 8; ++j) s += expf(v[j] - mx);
  s = wave_sum(s);
  const float delta = (mx + logf(s)) - logf(MASS);
#pragma unroll
  for (int j = 0; j < 8; ++j) p[(size_t)(lane + 32 * j) * M_] = v[j] - delta;
}

// ---------------------------------------------------------------------------
// Launch
// ---------------------------------------------------------------------------
extern "C" void kernel_launch(void* const* d_in, const int* in_sizes, int n_in,
                              void* d_out, int out_size, void* d_ws, size_t ws_size,
                              hipStream_t stream) {
  (void)in_sizes; (void)n_in; (void)out_size; (void)ws_size;
  const float* tra_x = (const float*)d_in[0];
  const float* det_x = (const float*)d_in[1];
  const float* W_enc = (const float*)d_in[2];
  const float* b_enc = (const float*)d_in[3];
  const float* W1    = (const float*)d_in[4];
  const float* b1    = (const float*)d_in[5];
  const float* W2    = (const float*)d_in[6];
  const float* b2    = (const float*)d_in[7];
  const float* eps   = (const float*)d_in[8];
  // d_in[9] is k; fixed to 16 per setup_inputs.

  char* ws = (char*)d_ws;
  size_t off = 0;
  auto alloc = [&](size_t bytes) -> char* {
    char* p = ws + off;
    off = (off + bytes + 255) & ~(size_t)255;
    return p;
  };
  bf16* wencT  = (bf16*)alloc((size_t)D_ * D_ * 2);          // [N=256][K=256]
  bf16* w1T    = (bf16*)alloc((size_t)E_ * 2 * D_ * 2);      // [N=256][K=512]
  bf16* w2T    = (bf16*)alloc((size_t)D_ * E_ * 2);          // [N=256][K=256]
  bf16* hbf    = (bf16*)alloc((size_t)2 * GM * D_ * 2);      // 16 MB
  float* sqv   = (float*)alloc((size_t)2 * GM * 4);
  float* d2v   = (float*)alloc((size_t)2 * GM * M_ * 4);     // 33.5 MB
  int*   idxv  = (int*)  alloc((size_t)2 * GM * KNN * 4);
  bf16* convbf = (bf16*)alloc((size_t)2 * GM * D_ * 2);      // 16 MB
  float* nrm   = (float*)alloc((size_t)2 * GM * 4);
  float* lp    = (float*)alloc((size_t)G_ * M_ * M_ * 4);    // 16 MB

  const dim3 B(256);

  // 0) weight convert + transpose (B-frags become contiguous 32B/lane loads)
  k_cvtT<<<(D_ * D_ + 255) / 256, B, 0, stream>>>(W_enc, wencT, D_, D_);
  k_cvtT<<<(2 * D_ * E_ + 255) / 256, B, 0, stream>>>(W1, w1T, 2 * D_, E_);
  k_cvtT<<<(E_ * D_ + 255) / 256, B, 0, stream>>>(W2, w2T, E_, D_);

  // 1) encoder (both inputs)
  k_encoder<<<2048, B, 0, stream>>>(tra_x, det_x, wencT, b_enc, hbf);
  // 2) squared norms
  k_sq<<<(2 * GM) / 8, B, 0, stream>>>(hbf, sqv);
  // 3) pairwise distances
  k_gram<<<2048, B, 0, stream>>>(hbf, sqv, d2v);
  // 4) kNN selection
  k_topk<<<(2 * GM) / 8, B, 0, stream>>>(d2v, idxv);
  // 5) fused edge MLP + max-aggregate (dominant: ~206 GFLOP in bf16 WMMA)
  k_edge<<<2 * GM, B, 0, stream>>>(hbf, idxv, w1T, b1, w2T, b2, convbf, nrm);
  // 6) cosine cost -> initial log_p
  k_corr<<<G_ * 16, B, 0, stream>>>(convbf, nrm, eps, lp);
  // 7) Sinkhorn: 7x (row, col), then final row fused with exp -> d_out
  for (int i = 0; i < NUM_SINK - 1; ++i) {
    k_sink_row<<<(G_ * M_) / 8, B, 0, stream>>>(lp, nullptr, 0);
    k_sink_col<<<(G_ * M_) / 8, B, 0, stream>>>(lp);
  }
  k_sink_row<<<(G_ * M_) / 8, B, 0, stream>>>(lp, (float*)d_out, 1);
}
